// classifier_attention_88845693485455
// MI455X (gfx1250) — matrix operand order repaired and checked by reading, not compiled, not run on hardware
//
#include <hip/hip_runtime.h>
#include <math.h>

typedef __attribute__((ext_vector_type(16))) __bf16 v16bf;
typedef __attribute__((ext_vector_type(8)))  __bf16 v8bf;
typedef __attribute__((ext_vector_type(4)))  __bf16 v4bf;
typedef __attribute__((ext_vector_type(8)))  float  v8f;
typedef __attribute__((ext_vector_type(4)))  float  v4f;

// Native converts: let clang emit v_cvt_pk_bf16_f32 / v_cvt_f32_bf16 on gfx1250
static __device__ __forceinline__ __bf16 f2bf(float f) { return (__bf16)f; }
static __device__ __forceinline__ float  bf2f(__bf16 b) { return (float)b; }

// Problem constants
constexpr int Dd = 256;          // feature dim
constexpr int Ll = 200;          // sequence length
constexpr int LT = 13;           // ceil(200/16) M-tiles
constexpr int WS = 264;          // padded LDS row stride (bf16 elems) for W
constexpr int HS = 264;          // padded LDS row stride (bf16 elems) for H

// LDS layout (bytes)
constexpr int W_OFF     = 0;
constexpr int H_OFF     = 256 * WS * 2;          // 135168
constexpr int BIAS_OFF  = H_OFF + 208 * HS * 2;  // 244992
constexpr int ACT_OFF   = BIAS_OFF + 1024;       // 246016
constexpr int SCORE_OFF = ACT_OFF + 1024;        // 247040
constexpr int RED_OFF   = SCORE_OFF + 1024;      // 248064
constexpr int SMEM_BYTES = RED_OFF + 1024;       // 249088 (< 320 KB)

__global__ __launch_bounds__(256, 1)
void branch_attn_kernel(const float* __restrict__ Mx, const float* __restrict__ Nx,
                        const float* __restrict__ act,
                        const float* __restrict__ Wm, const float* __restrict__ bm,
                        const float* __restrict__ Wn, const float* __restrict__ bn,
                        float* __restrict__ out)
{
  extern __shared__ char smem[];
  __bf16* Wlds    = (__bf16*)(smem + W_OFF);
  __bf16* Hlds    = (__bf16*)(smem + H_OFF);
  float*  biasLds = (float*)(smem + BIAS_OFF);
  float*  actLds  = (float*)(smem + ACT_OFF);
  float*  scoreLds= (float*)(smem + SCORE_OFF);
  float*  redLds  = (float*)(smem + RED_OFF);

  const int t     = threadIdx.x;
  const int batch = blockIdx.x >> 1;
  const int br    = blockIdx.x & 1;
  const float* X  = br ? Nx : Mx;
  const float* Wg = br ? Wn : Wm;
  const float* bg = br ? bn : bm;

  // ---- Phase 0: stage W (fp32 -> bf16), bias, act_r into LDS ----
  {
    const v4f* W4 = (const v4f*)Wg;
    #pragma unroll 4
    for (int j = 0; j < 64; ++j) {
      int f = j * 256 + t;            // float4 index over 256x256 W
      int n = f >> 6;                 // row (output feature)
      int k = (f & 63) << 2;          // col (input feature)
      v4f w = W4[f];
      v4bf wb;
      #pragma unroll
      for (int e2 = 0; e2 < 4; ++e2) wb[e2] = f2bf(w[e2]);
      *(v4bf*)(Wlds + n * WS + k) = wb;
    }
    biasLds[t] = bg[t];
    actLds[t]  = act[(size_t)batch * Dd + t];
  }
  __syncthreads();

  // ---- Phase 1: H = relu(X @ W^T + b) via v_wmma_f32_16x16x32_bf16 ----
  const int wave = t >> 5;
  const int lane = t & 31;
  const int half = lane >> 4;     // 0 or 1
  const int l16  = lane & 15;
  const int koffA = half << 3;    // A-frag K offset (interleaved layout)
  const int koffB = half << 3;    // B-frag K offset (interleaved layout)

  for (int mt = wave; mt < LT; mt += 8) {
    v8f acc[16];
    #pragma unroll
    for (int i = 0; i < 16; ++i) {
      #pragma unroll
      for (int e2 = 0; e2 < 8; ++e2) acc[i][e2] = 0.0f;
    }

    int row = mt * 16 + l16;
    if (row > Ll - 1) row = Ll - 1;   // clamp pad rows (masked later via att=0)
    const float* xr = X + (size_t)batch * (size_t)(Ll * Dd) + (size_t)row * Dd;

    for (int kt = 0; kt < 8; ++kt) {
      // A fragment: lanes 0-15 hold K={0..7,16..23}, lanes 16-31 K={8..15,24..31}
      const int k0 = kt * 32 + koffA;
      v4f a0 = *(const v4f*)(xr + k0);
      v4f a1 = *(const v4f*)(xr + k0 + 4);
      v4f a2 = *(const v4f*)(xr + k0 + 16);
      v4f a3 = *(const v4f*)(xr + k0 + 20);
      v16bf afrag;
      #pragma unroll
      for (int e2 = 0; e2 < 4; ++e2) {
        afrag[e2]      = f2bf(a0[e2]);
        afrag[4 + e2]  = f2bf(a1[e2]);
        afrag[8 + e2]  = f2bf(a2[e2]);
        afrag[12 + e2] = f2bf(a3[e2]);
      }
      const int kb = kt * 32 + koffB;   // B: lanes 0-15 K={0..7,16..23}, lanes 16-31 K={8..15,24..31}
      #pragma unroll
      for (int nt = 0; nt < 16; ++nt) {
        const __bf16* wr = Wlds + (nt * 16 + l16) * WS + kb;
        v8bf b0 = *(const v8bf*)(wr);
        v8bf b1 = *(const v8bf*)(wr + 16);
        v16bf bfrag;
        #pragma unroll
        for (int e2 = 0; e2 < 8; ++e2) { bfrag[e2] = b0[e2]; bfrag[8 + e2] = b1[e2]; }
        acc[nt] = __builtin_amdgcn_wmma_f32_16x16x32_bf16(
            false, afrag, false, bfrag, (short)0, acc[nt], false, false);
      }
    }

    // Epilogue: bias + ReLU, store H as bf16 to LDS
    #pragma unroll
    for (int nt = 0; nt < 16; ++nt) {
      const int n = nt * 16 + l16;
      const float bv = biasLds[n];
      #pragma unroll
      for (int i = 0; i < 8; ++i) {
        const int r = mt * 16 + i + (half << 3);   // C layout: VGPR i -> M=i / i+8
        float h = acc[nt][i] + bv;
        h = h > 0.0f ? h : 0.0f;
        Hlds[r * HS + n] = f2bf(h);
      }
    }
  }
  __syncthreads();

  // ---- Phase 2: scores[r] = H[r,:] . act_r ; softmax over L ----
  {
    float s = -INFINITY;
    if (t < 208) {
      float acc0 = 0.0f;
      const __bf16* hr = Hlds + t * HS;   // row base: t*528 B, 16B aligned
      #pragma unroll 4
      for (int kb8 = 0; kb8 < 32; ++kb8) {
        v8bf hv = *(const v8bf*)(hr + kb8 * 8);     // ds_load_b128
        #pragma unroll
        for (int e2 = 0; e2 < 8; ++e2)
          acc0 += bf2f(hv[e2]) * actLds[kb8 * 8 + e2];
      }
      s = (t < Ll) ? acc0 : -INFINITY;
    }
    scoreLds[t] = s;
    redLds[t]   = s;
  }
  __syncthreads();
  for (int sft = 128; sft > 0; sft >>= 1) {
    if (t < sft) redLds[t] = fmaxf(redLds[t], redLds[t + sft]);
    __syncthreads();
  }
  const float mx = redLds[0];
  __syncthreads();
  const float ev = __expf(scoreLds[t] - mx);   // exp(-inf) = 0 for pads
  redLds[t] = ev;
  __syncthreads();
  for (int sft = 128; sft > 0; sft >>= 1) {
    if (t < sft) redLds[t] += redLds[t + sft];
    __syncthreads();
  }
  const float inv = 1.0f / (redLds[0] * (float)Ll);   // fold the /L here
  __syncthreads();
  scoreLds[t] = ev * inv;                              // att weights
  __syncthreads();

  // ---- Phase 3: out[n] = sum_r att[r] * H[r][n] ----
  {
    float o = 0.0f;
    for (int r = 0; r < Ll; ++r) o += scoreLds[r] * bf2f(Hlds[r * HS + t]);
    out[(size_t)batch * (2 * Dd) + (size_t)br * Dd + t] = o;
  }
}

extern "C" void kernel_launch(void* const* d_in, const int* in_sizes, int n_in,
                              void* d_out, int out_size, void* d_ws, size_t ws_size,
                              hipStream_t stream) {
  (void)in_sizes; (void)n_in; (void)out_size; (void)d_ws; (void)ws_size;
  const float* M  = (const float*)d_in[0];
  const float* N  = (const float*)d_in[1];
  const float* ar = (const float*)d_in[2];
  const float* Wm = (const float*)d_in[3];
  const float* bm = (const float*)d_in[4];
  const float* Wn = (const float*)d_in[5];
  const float* bn = (const float*)d_in[6];
  float* out = (float*)d_out;

  dim3 grid(2048);   // (batch, branch) pairs: 1024 * 2
  dim3 block(256);   // 8 waves of 32
  branch_attn_kernel<<<grid, block, SMEM_BYTES, stream>>>(M, N, ar, Wm, bm, Wn, bn, out);
}
